// ScaledTernaryLinear_49065706389534
// MI455X (gfx1250) — matrix operand order repaired and checked by reading, not compiled, not run on hardware
//
#include <hip/hip_runtime.h>
#include <hip/hip_bf16.h>
#include <cstdint>
#include <cstddef>

// ---------------------------------------------------------------------------
// ScaledTernaryLinear for MI455X (gfx1250, wave32, WMMA)
//
//   y = (x @ ternary(W)^T) * scale + bias
//   x: [4096, 4096] f32, W: [16384, 4096] f32, out: [4096, 16384] f32
//
// Compute-bound GEMM (~550 GFLOP, AI ~900 FLOP/B):
//   1) Prologue: quantize W -> ternary f16 (exact {-1,0,+1}) into d_ws.
//      128 MB of f16 weights fit the 192 MB L2 -> W re-reads are L2 traffic.
//   2) Prologue: convert x -> f16 (32 MB) into d_ws (~3e-4 rel elem error;
//      f32 WMMA accumulation keeps the K=4096 reduction accurate).
//   3) GEMM with v_wmma_f32_16x16x32_f16: 8-wave blocks, 128x256 block tile,
//      64x64 per-wave tile (16 WMMA per K-step). __launch_bounds__(256,1)
//      lifts the VGPR budget so all 8 fragments occupy distinct registers
//      (stepped s_wait_loadcnt instead of full drains); global_prefetch_b8
//      stream prefetch; NON-TEMPORAL epilogue stores so the 256 MB output
//      stream doesn't evict the L2-resident weights; fused scale/bias.
// ---------------------------------------------------------------------------

typedef __attribute__((ext_vector_type(4)))  _Float16 v4h;
typedef __attribute__((ext_vector_type(8)))  _Float16 v8h;
typedef __attribute__((ext_vector_type(16))) _Float16 v16h;
typedef __attribute__((ext_vector_type(8)))  float    v8f;

#define K_DIM 4096
#define M_DIM 4096
#define N_DIM 16384
#define TERN_THRESH 0.05f

// ---------------- Prologue 1: W f32 -> ternary f16 -------------------------
__global__ __launch_bounds__(256)
void quantize_w_kernel(const float* __restrict__ w, _Float16* __restrict__ wq) {
    const size_t i = ((size_t)blockIdx.x * blockDim.x + threadIdx.x);
    const float4 v = ((const float4*)w)[i];
    v4h o;
    o[0] = (_Float16)((v.x > TERN_THRESH) ? 1.0f : (v.x < -TERN_THRESH ? -1.0f : 0.0f));
    o[1] = (_Float16)((v.y > TERN_THRESH) ? 1.0f : (v.y < -TERN_THRESH ? -1.0f : 0.0f));
    o[2] = (_Float16)((v.z > TERN_THRESH) ? 1.0f : (v.z < -TERN_THRESH ? -1.0f : 0.0f));
    o[3] = (_Float16)((v.w > TERN_THRESH) ? 1.0f : (v.w < -TERN_THRESH ? -1.0f : 0.0f));
    ((v4h*)wq)[i] = o;
}

// ---------------- Prologue 2: x f32 -> f16 ---------------------------------
__global__ __launch_bounds__(256)
void convert_x_kernel(const float* __restrict__ x, _Float16* __restrict__ xq) {
    const size_t i = ((size_t)blockIdx.x * blockDim.x + threadIdx.x);
    const float4 v = ((const float4*)x)[i];
    v4h o;
    o[0] = (_Float16)v.x;
    o[1] = (_Float16)v.y;
    o[2] = (_Float16)v.z;
    o[3] = (_Float16)v.w;
    ((v4h*)xq)[i] = o;
}

// ---------------- Main WMMA GEMM -------------------------------------------
// Block = 256 threads = 8 waves. Block tile: 128 (M) x 256 (N).
// Wave grid: 2 waves along M x 4 waves along N; each wave owns 64x64 =
// a 4x4 grid of 16x16 WMMA tiles (128 accumulator VGPRs).
// __launch_bounds__(256, 1): allow the register allocator to exceed the
// default occupancy-clamped 192-VGPR budget so every fragment gets its own
// registers (removes forced s_wait_loadcnt 0x0 drains mid-iteration).
__global__ __launch_bounds__(256, 1)
void ternary_gemm_wmma(const _Float16* __restrict__ xq,
                       const _Float16* __restrict__ wq,
                       const float*    __restrict__ scale,
                       const float*    __restrict__ bias,
                       float*          __restrict__ out) {
    const int lane = threadIdx.x & 31;
    const int wave = threadIdx.x >> 5;
    const int wm   = wave & 1;   // 0..1 -> 64-row slice
    const int wn   = wave >> 1;  // 0..3 -> 64-col slice

    const int m_base = blockIdx.y * 128 + wm * 64;   // wave's first output row
    const int n_base = blockIdx.x * 256 + wn * 64;   // wave's first output col

    const int row   = lane & 15;            // M (for A) / N (for B) within tile
    const int hi16  = lane >> 4;            // lane group 0/1
    const int koffA = hi16 << 3;            // A: 0 or 8   (chunks at +0, +16)
    const int koffB = hi16 << 3;            // B: 0 or 8   (chunks at +0, +16)

    // A fragment source rows (x f16, row-major over K).
    // 16-bit A 16x32 layout: lane<16 holds K {0..7,16..23}, lane>=16 {8..15,24..31}.
    const _Float16* aPtr[4];
    aPtr[0] = xq + (size_t)(m_base + row) * K_DIM + koffA;
#pragma unroll
    for (int mi = 1; mi < 4; ++mi) aPtr[mi] = aPtr[mi - 1] + (size_t)16 * K_DIM;

    // B fragment source rows (ternary W f16; B column n == W row n, contiguous K).
    // 16-bit B 32x16 layout: lane<16 holds K {0..7,16..23}, lane>=16 {8..15,24..31}.
    const _Float16* bPtr[4];
#pragma unroll
    for (int ni = 0; ni < 4; ++ni)
        bPtr[ni] = wq + (size_t)(n_base + ni * 16 + row) * K_DIM + koffB;

    v8f acc[4][4];
#pragma unroll
    for (int mi = 0; mi < 4; ++mi)
#pragma unroll
        for (int ni = 0; ni < 4; ++ni)
            acc[mi][ni] = (v8f){};

    for (int k0 = 0; k0 < K_DIM; k0 += 32) {
        // Stream prefetch ~8 K-iterations ahead (global_prefetch_b8)
        if (k0 + 256 < K_DIM) {
#pragma unroll
            for (int mi = 0; mi < 4; ++mi) __builtin_prefetch(aPtr[mi] + k0 + 256);
#pragma unroll
            for (int ni = 0; ni < 4; ++ni) __builtin_prefetch(bPtr[ni] + k0 + 256);
        }

        // Load ALL fragments for this K-step up front, into distinct registers,
        // so the scheduler can issue one load clause and step s_wait_loadcnt
        // down across the 16-WMMA stream (no full drains mid-iteration).
        v16h a[4], b[4];
#pragma unroll
        for (int ni = 0; ni < 4; ++ni) {
            v8h lo = *(const v8h*)(bPtr[ni] + k0);
            v8h hi = *(const v8h*)(bPtr[ni] + k0 + 16);
            b[ni] = __builtin_shufflevector(lo, hi,
                                            0, 1, 2, 3, 4, 5, 6, 7,
                                            8, 9, 10, 11, 12, 13, 14, 15);
        }
#pragma unroll
        for (int mi = 0; mi < 4; ++mi) {
            v8h lo = *(const v8h*)(aPtr[mi] + k0);
            v8h hi = *(const v8h*)(aPtr[mi] + k0 + 16);
            a[mi] = __builtin_shufflevector(lo, hi,
                                            0, 1, 2, 3, 4, 5, 6, 7,
                                            8, 9, 10, 11, 12, 13, 14, 15);
        }

#pragma unroll
        for (int mi = 0; mi < 4; ++mi)
#pragma unroll
            for (int ni = 0; ni < 4; ++ni)
                acc[mi][ni] = __builtin_amdgcn_wmma_f32_16x16x32_f16(
                    /*neg_a=*/false, a[mi],
                    /*neg_b=*/false, b[ni],
                    /*c_mod=*/(short)0, acc[mi][ni],
                    /*reuse_a=*/false, /*reuse_b=*/false);
    }

    // Epilogue: D 16x16 f32 layout -> VGPR r holds row (r + 8*hi16), lane%16 = col.
    // Non-temporal stores: the 256 MB output stream must not evict the
    // L2-resident 128 MB weight array.
#pragma unroll
    for (int ni = 0; ni < 4; ++ni) {
        const int   n_g = n_base + ni * 16 + row;
        const float sc  = scale[n_g];
        const float bi  = bias[n_g];
#pragma unroll
        for (int mi = 0; mi < 4; ++mi) {
#pragma unroll
            for (int r = 0; r < 8; ++r) {
                const int m_g = m_base + mi * 16 + r + (hi16 << 3);
                const float v = acc[mi][ni][r] * sc + bi;
                __builtin_nontemporal_store(v, &out[(size_t)m_g * N_DIM + n_g]);
            }
        }
    }
}

// ---------------------------------------------------------------------------
extern "C" void kernel_launch(void* const* d_in, const int* in_sizes, int n_in,
                              void* d_out, int out_size, void* d_ws, size_t ws_size,
                              hipStream_t stream) {
    const float* x     = (const float*)d_in[0];
    const float* w     = (const float*)d_in[1];
    const float* scale = (const float*)d_in[2];
    const float* bias  = (const float*)d_in[3];
    float*       out   = (float*)d_out;

    // Workspace layout: [ Wq f16 : 128 MB | xq f16 : 32 MB ]  (needs 160 MB)
    _Float16* wq = (_Float16*)d_ws;
    _Float16* xq = (_Float16*)((char*)d_ws + (size_t)N_DIM * K_DIM * sizeof(_Float16));

    {
        const int n4 = (N_DIM * K_DIM) / 4;           // 16,777,216 float4s
        quantize_w_kernel<<<n4 / 256, 256, 0, stream>>>(w, wq);
    }
    {
        const int n4 = (M_DIM * K_DIM) / 4;           // 4,194,304 float4s
        convert_x_kernel<<<n4 / 256, 256, 0, stream>>>(x, xq);
    }

    dim3 grid(N_DIM / 256, M_DIM / 128);              // 64 x 32 blocks
    ternary_gemm_wmma<<<grid, 256, 0, stream>>>(xq, wq, scale, bias, out);
}
